// PeepholeLSTM_80848464380268
// MI455X (gfx1250) — matrix operand order repaired and checked by reading, not compiled, not run on hardware
//
#include <hip/hip_runtime.h>

// Problem constants (match reference)
#define DIMN 1024
#define HIDN 2048
#define CATN 4096   // DIM + DIM + HID
#define LENN 2048
#define BATN 16

#define GSPLIT 8    // K-split (waves per tile) in gate kernel: 4096/32/8 = 16 ksteps/wave
#define OSPLIT 8    // K-split in out kernel: 2048/32/8 = 8 ksteps/wave

typedef __attribute__((ext_vector_type(16))) __bf16 v16bf;
typedef __attribute__((ext_vector_type(8)))  __bf16 v8bf;
typedef __attribute__((ext_vector_type(8)))  float  v8f;

union ABfrag { v16bf v; v8bf h[2]; };

__device__ __forceinline__ unsigned short f2bf(float f) {
    unsigned u = __float_as_uint(f);
    u += 0x7FFFu + ((u >> 16) & 1u);   // round-to-nearest-even
    return (unsigned short)(u >> 16);
}

__device__ __forceinline__ v8f wmma_bf16(v16bf a, v16bf b, v8f c) {
    // (neg_a, A, neg_b, B, c_mod, C, reuse_a, reuse_b)
    return __builtin_amdgcn_wmma_f32_16x16x32_bf16(false, a, false, b, (short)0, c, false, false);
}

__device__ __forceinline__ float fast_sig(float z) {           // 1/(1+e^-z), v_rcp_f32
    return __builtin_amdgcn_rcpf(1.f + __expf(-z));
}

// ---------------------------------------------------------------------------
// Pack fp32 weight [N,K] row-major -> bf16 WMMA B-panel layout:
//   per 16-col tile, per 32-row k-step: 32 lanes x 16 contiguous bf16.
//   lane = (n%16) + 16*((kk/8)%2), elem i = kk%8 + 8*(kk/16)  (kk = k%32)
// ---------------------------------------------------------------------------
__global__ void pack_kernel(const float* __restrict__ W, unsigned short* __restrict__ P,
                            int N, int K) {
    int tid = blockIdx.x * blockDim.x + threadIdx.x;
    if (tid >= N * K) return;
    int n = tid / K, k = tid - n * K;
    int tile = n >> 4, ks = k >> 5, kk = k & 31;
    int lane = (n & 15) | (((kk >> 3) & 1) << 4);
    int i = (kk & 7) | ((kk >> 4) << 3);
    P[(((size_t)tile * (K >> 5) + ks) * 32 + lane) * 16 + i] = f2bf(W[tid]);
}

// negA[n] = -exp(ln_forget_a[n])  (precompute once per call)
__global__ void prep_kernel(const float* __restrict__ lna, float* __restrict__ negA) {
    int n = blockIdx.x * blockDim.x + threadIdx.x;
    if (n < HIDN) negA[n] = -__expf(lna[n]);
}

// Build cat0 = [x(:,0,:), o0, h0] (bf16) and h state (f32, broadcast h0)
__global__ void init_kernel(const float* __restrict__ x, const float* __restrict__ h0,
                            const float* __restrict__ o0,
                            unsigned short* __restrict__ catA, float* __restrict__ h) {
    int tid = blockIdx.x * blockDim.x + threadIdx.x;   // 0 .. 16*4096-1
    int m = tid >> 12;
    int c = tid & 4095;
    float v;
    if (c < DIMN)            v = x[(size_t)m * LENN * DIMN + c];
    else if (c < 2 * DIMN)   v = o0[c - DIMN];
    else { v = h0[c - 2 * DIMN]; h[m * HIDN + (c - 2 * DIMN)] = v; }
    catA[tid] = f2bf(v);
}

// ---------------------------------------------------------------------------
// Per-step kernel 1: forget/input gates + hidden update + Wo1 branch.
// Block = 8 waves = 1 tile x 8 K-parts. 192 blocks -> 1536 waves.
//   tiles 0..127 : Wf/Wi1/Wi2 GEMM tile over CAT, elementwise nh
//   tiles 128..191: Wo1 GEMM tile, silu
// Split-K partials reduced through LDS; kpart==0 wave keeps its own partials
// in registers and does the tail.
// ---------------------------------------------------------------------------
__global__ __launch_bounds__(256) void gate_kernel(
    const unsigned short* __restrict__ catIn,
    const unsigned short* __restrict__ WfP,
    const unsigned short* __restrict__ Wi1P,
    const unsigned short* __restrict__ Wi2P,
    const unsigned short* __restrict__ Wo1P,
    const float* __restrict__ bfv,  const float* __restrict__ negA,
    const float* __restrict__ bi1,  const float* __restrict__ bi2,
    const float* __restrict__ bo1,
    float* __restrict__ h, float* __restrict__ o1buf,
    unsigned short* __restrict__ catOut)
{
    __shared__ float red[GSPLIT][3][256];            // [kpart][gemm][j*32+lane]
    const int lane  = threadIdx.x & 31;
    const int kpart = threadIdx.x >> 5;              // 0..7
    const int tile  = blockIdx.x;                    // 0..191
    const int m16  = lane & 15;
    const int half = lane >> 4;
    const bool isHid = tile < HIDN / 16;

    // A fragment rows: lane holds row m16; K = ks*32 + half*8 + {0..7, 16..23}
    const unsigned short* arow = catIn + m16 * CATN + half * 8;
    const int ks0 = kpart * (CATN / 32 / GSPLIT);
    v8f a0 = {}, a1 = {}, a2 = {};

    if (isHid) {
        const size_t tb = (size_t)tile * (CATN / 32) * 512 + lane * 16;
        const unsigned short* pF  = WfP  + tb;
        const unsigned short* pI1 = Wi1P + tb;
        const unsigned short* pI2 = Wi2P + tb;
        #pragma unroll 2
        for (int ks = ks0; ks < ks0 + CATN / 32 / GSPLIT; ++ks) {
            ABfrag a;
            a.h[0] = *(const v8bf*)(arow + ks * 32);
            a.h[1] = *(const v8bf*)(arow + ks * 32 + 16);
            v16bf bF  = *(const v16bf*)(pF  + ks * 512);
            v16bf bI1 = *(const v16bf*)(pI1 + ks * 512);
            v16bf bI2 = *(const v16bf*)(pI2 + ks * 512);
            a0 = wmma_bf16(a.v, bF,  a0);
            a1 = wmma_bf16(a.v, bI1, a1);
            a2 = wmma_bf16(a.v, bI2, a2);
        }
    } else {
        const size_t tb = (size_t)(tile - HIDN / 16) * (CATN / 32) * 512 + lane * 16;
        const unsigned short* pO = Wo1P + tb;
        #pragma unroll 2
        for (int ks = ks0; ks < ks0 + CATN / 32 / GSPLIT; ++ks) {
            ABfrag a;
            a.h[0] = *(const v8bf*)(arow + ks * 32);
            a.h[1] = *(const v8bf*)(arow + ks * 32 + 16);
            v16bf b = *(const v16bf*)(pO + ks * 512);
            a0 = wmma_bf16(a.v, b, a0);
        }
    }

    // deposit split-K partials (conflict-free: lane-contiguous); wave 0 keeps regs
    if (kpart != 0) {
        #pragma unroll
        for (int j = 0; j < 8; ++j) red[kpart][0][j * 32 + lane] = a0[j];
        if (isHid) {
            #pragma unroll
            for (int j = 0; j < 8; ++j) red[kpart][1][j * 32 + lane] = a1[j];
            #pragma unroll
            for (int j = 0; j < 8; ++j) red[kpart][2][j * 32 + lane] = a2[j];
        }
    }
    __syncthreads();
    if (kpart != 0) return;

    if (isHid) {
        const int n = tile * 16 + m16;
        const float bF = bfv[n], nA = negA[n], b1 = bi1[n], b2 = bi2[n];
        #pragma unroll
        for (int j = 0; j < 8; ++j) {
            const int idx = j * 32 + lane;
            float zf = a0[j] + bF;
            float z1 = a1[j] + b1;
            float z2 = a2[j] + b2;
            #pragma unroll
            for (int p = 1; p < GSPLIT; ++p) {
                zf += red[p][0][idx];
                z1 += red[p][1][idx];
                z2 += red[p][2][idx];
            }
            float sp = (zf > 20.f) ? zf : __logf(1.f + __expf(zf));   // softplus
            float forget = __expf(nA * sp);
            float inp = z1 * fast_sig(z1) * z2;                       // silu * lin
            const int m = j + half * 8;                               // C/D layout
            float nh = h[m * HIDN + n] * forget + inp;
            h[m * HIDN + n] = nh;
            catOut[m * CATN + 2 * DIMN + n] = f2bf(nh);
        }
    } else {
        const int n = (tile - HIDN / 16) * 16 + m16;
        const float b = bo1[n];
        #pragma unroll
        for (int j = 0; j < 8; ++j) {
            const int idx = j * 32 + lane;
            float z = a0[j] + b;
            #pragma unroll
            for (int p = 1; p < GSPLIT; ++p) z += red[p][0][idx];
            const int m = j + half * 8;
            o1buf[m * DIMN + n] = z * fast_sig(z);                    // silu
        }
    }
}

// ---------------------------------------------------------------------------
// Per-step kernel 2: out = silu(cat@Wo1+bo1) * (nh@Wo2 + bo2);
// write output slice, stage out + next x into cat buffer.
// Block = 8 waves = 1 tile x 8 K-parts. 64 blocks -> 512 waves.
// ---------------------------------------------------------------------------
__global__ __launch_bounds__(256) void out_kernel(
    unsigned short* __restrict__ catN,        // nh region read; x/out regions written
    const unsigned short* __restrict__ Wo2P,
    const float* __restrict__ bo2,
    const float* __restrict__ o1buf,
    const float* __restrict__ x,
    float* __restrict__ outp,
    int t)
{
    __shared__ float red[OSPLIT][256];
    const int lane  = threadIdx.x & 31;
    const int kpart = threadIdx.x >> 5;              // 0..7
    const int tile  = blockIdx.x;                    // 0..63
    const int m16  = lane & 15;
    const int half = lane >> 4;

    const unsigned short* arow = catN + m16 * CATN + 2 * DIMN + half * 8;  // nh (bf16)
    const unsigned short* pO = Wo2P + (size_t)tile * (HIDN / 32) * 512 + lane * 16;
    const int ks0 = kpart * (HIDN / 32 / OSPLIT);
    v8f acc = {};
    #pragma unroll 2
    for (int ks = ks0; ks < ks0 + HIDN / 32 / OSPLIT; ++ks) {
        ABfrag a;
        a.h[0] = *(const v8bf*)(arow + ks * 32);
        a.h[1] = *(const v8bf*)(arow + ks * 32 + 16);
        v16bf b = *(const v16bf*)(pO + ks * 512);
        acc = wmma_bf16(a.v, b, acc);
    }
    if (kpart != 0) {
        #pragma unroll
        for (int j = 0; j < 8; ++j) red[kpart][j * 32 + lane] = acc[j];
    }
    __syncthreads();

    if (kpart == 0) {
        const int n = tile * 16 + m16;
        const float b2 = bo2[n];
        #pragma unroll
        for (int j = 0; j < 8; ++j) {
            const int idx = j * 32 + lane;
            float z = acc[j] + b2;
            #pragma unroll
            for (int p = 1; p < OSPLIT; ++p) z += red[p][idx];
            const int m = j + half * 8;
            float ov = o1buf[m * DIMN + n] * z;
            outp[((size_t)m * LENN + t) * DIMN + n] = ov;
            catN[m * CATN + DIMN + n] = f2bf(ov);
        }
    }
    // stage x_{t+1} into cat (16*1024 elems over 64*256 = 16384 threads: 1 each)
    if (t + 1 < LENN) {
        const int e = blockIdx.x * blockDim.x + threadIdx.x;   // 0..16383
        const int m = e >> 10, n2 = e & 1023;
        catN[m * CATN + n2] = f2bf(x[((size_t)m * LENN + (t + 1)) * DIMN + n2]);
    }
}

// ---------------------------------------------------------------------------
extern "C" void kernel_launch(void* const* d_in, const int* in_sizes, int n_in,
                              void* d_out, int out_size, void* d_ws, size_t ws_size,
                              hipStream_t stream) {
    const float* x    = (const float*)d_in[0];
    const float* Wf   = (const float*)d_in[1];
    const float* bfv  = (const float*)d_in[2];
    const float* lna  = (const float*)d_in[3];
    const float* Wi1  = (const float*)d_in[4];
    const float* bi1  = (const float*)d_in[5];
    const float* Wi2  = (const float*)d_in[6];
    const float* bi2  = (const float*)d_in[7];
    const float* Wo1  = (const float*)d_in[8];
    const float* bo1  = (const float*)d_in[9];
    const float* Wo2  = (const float*)d_in[10];
    const float* bo2  = (const float*)d_in[11];
    const float* h0   = (const float*)d_in[12];
    const float* o0   = (const float*)d_in[13];

    char* w = (char*)d_ws;
    unsigned short* WfP  = (unsigned short*)w;  w += (size_t)HIDN * CATN * 2;
    unsigned short* Wi1P = (unsigned short*)w;  w += (size_t)HIDN * CATN * 2;
    unsigned short* Wi2P = (unsigned short*)w;  w += (size_t)HIDN * CATN * 2;
    unsigned short* Wo1P = (unsigned short*)w;  w += (size_t)DIMN * CATN * 2;
    unsigned short* Wo2P = (unsigned short*)w;  w += (size_t)DIMN * HIDN * 2;
    unsigned short* catA = (unsigned short*)w;  w += (size_t)BATN * CATN * 2;
    unsigned short* catB = (unsigned short*)w;  w += (size_t)BATN * CATN * 2;
    float* hbuf  = (float*)w;                   w += (size_t)BATN * HIDN * 4;
    float* o1buf = (float*)w;                   w += (size_t)BATN * DIMN * 4;
    float* negA  = (float*)w;                   w += (size_t)HIDN * 4;

    // One-time per call: pack weights to bf16 WMMA panels, precompute, init state
    pack_kernel<<<(HIDN * CATN + 255) / 256, 256, 0, stream>>>(Wf,  WfP,  HIDN, CATN);
    pack_kernel<<<(HIDN * CATN + 255) / 256, 256, 0, stream>>>(Wi1, Wi1P, HIDN, CATN);
    pack_kernel<<<(HIDN * CATN + 255) / 256, 256, 0, stream>>>(Wi2, Wi2P, HIDN, CATN);
    pack_kernel<<<(DIMN * CATN + 255) / 256, 256, 0, stream>>>(Wo1, Wo1P, DIMN, CATN);
    pack_kernel<<<(DIMN * HIDN + 255) / 256, 256, 0, stream>>>(Wo2, Wo2P, DIMN, HIDN);
    prep_kernel<<<HIDN / 256, 256, 0, stream>>>(lna, negA);
    init_kernel<<<(BATN * CATN) / 256, 256, 0, stream>>>(x, h0, o0, catA, hbuf);

    for (int t = 0; t < LENN; ++t) {
        unsigned short* cIn  = (t & 1) ? catB : catA;
        unsigned short* cOut = (t & 1) ? catA : catB;
        // 192 tiles x 8 K-parts = 1536 waves = 192 blocks * 8 waves
        gate_kernel<<<192, 256, 0, stream>>>(cIn, WfP, Wi1P, Wi2P, Wo1P,
                                             bfv, negA, bi1, bi2, bo1,
                                             hbuf, o1buf, cOut);
        // 64 tiles x 8 K-parts = 512 waves = 64 blocks * 8 waves
        out_kernel<<<64, 256, 0, stream>>>(cOut, Wo2P, bo2, o1buf, x, (float*)d_out, t);
    }
}
